// ImprovedSpatialGNN_48601849921796
// MI455X (gfx1250) — matrix operand order repaired and checked by reading, not compiled, not run on hardware
//
#include <hip/hip_runtime.h>
#include <hip/hip_bf16.h>
#include <math.h>

// ---------------- problem constants (match reference) ----------------
#define NN      50000          // nodes
#define NE      800000         // edges (before self loops)
#define ET      (NE + NN)      // edges incl self loops
#define FIN     128
#define HEADS   4
#define CH      64             // per-head channels
#define HC      (HEADS*CH)     // 256
#define BG      64             // graphs
#define MLPH    256
#define MLPH2   128
#define NGLOB   10000
#define NEG     0.2f
#define EPSBN   1e-5f

// ---------------- WMMA types ----------------
typedef __attribute__((ext_vector_type(16))) __bf16         v16bf;
typedef __attribute__((ext_vector_type(16))) unsigned short v16us;
typedef __attribute__((ext_vector_type(8)))  unsigned short v8us;
typedef __attribute__((ext_vector_type(8)))  float          v8f;

__device__ __forceinline__ unsigned short f32_to_bf16_rne(float f) {
  union { float f; unsigned u; } v; v.f = f;
  unsigned r = v.u + 0x7FFFu + ((v.u >> 16) & 1u);
  return (unsigned short)(r >> 16);
}

__device__ __forceinline__ void atomMaxF(float* p, float v) {
  __hip_atomic_fetch_max(p, v, __ATOMIC_RELAXED, __HIP_MEMORY_SCOPE_AGENT);
}

// ---------------- generic fill ----------------
__global__ void fill_kernel(float* p, long n, float v) {
  long i = (long)blockIdx.x * blockDim.x + threadIdx.x;
  if (i < n) p[i] = v;
}

// ---------------- f32 -> bf16 bulk convert (A operands) ----------------
__global__ void cvt_bf16_kernel(const float* __restrict__ x, unsigned short* __restrict__ y,
                                long n) {
  long i = (long)blockIdx.x * blockDim.x + threadIdx.x;
  if (i < n) y[i] = f32_to_bf16_rne(x[i]);
}

// ---------------- pack B[K,N] f32 -> WMMA fragment layout bf16 ----------------
// Bp[((t*ntiles + nt)*32 + lane)*16 + j] = bf16(B[t*32 + (lane>>4)*16 + j, nt*16 + (lane&15)])
__global__ void pack_B_kernel(const float* __restrict__ B, unsigned short* __restrict__ Bp,
                              int K, int N) {
  long total = (long)K * N;
  long i = (long)blockIdx.x * blockDim.x + threadIdx.x;
  if (i >= total) return;
  int ntiles = N >> 4;
  int  j  = (int)(i & 15);
  int  l  = (int)((i >> 4) & 31);
  long r  = i >> 9;
  int  nt = (int)(r % ntiles);
  int  t  = (int)(r / ntiles);
  int  k  = (t << 5) + ((l >> 4) << 4) + j;
  int  n  = (nt << 4) + (l & 15);
  Bp[i] = f32_to_bf16_rne(B[(size_t)k * N + n]);
}

// ---------------- WMMA GEMM: C[M,N] = A[M,K] @ B[K,N] (+bias[N]) ----------------
// A: bf16 row-major. Bp: pre-packed fragments. One wave computes a 16x(16*Q) C
// strip (Q n-tiles, compile-time) so each A fragment feeds Q WMMAs with zero
// control flow in the hot loop. M%16==0, N%16==0, K%64==0.
template <int Q>
__global__ __launch_bounds__(32)
void gemm_bf16_wmma(const unsigned short* __restrict__ A,
                    const unsigned short* __restrict__ Bp,
                    const float* __restrict__ bias, float* __restrict__ C,
                    int M, int K, int N, int ntile_base) {
  const int l      = threadIdx.x;      // 0..31
  const int lane   = l & 15;
  const int half   = l >> 4;           // 0 or 1
  const int m0     = blockIdx.x << 4;
  const int ntiles = N >> 4;
  const int nt0    = ntile_base + blockIdx.y * Q;
  const int row    = m0 + lane;

  v8f acc[Q] = {};
  for (int k0 = 0; k0 < K; k0 += 64) {        // K is a multiple of 64 for all calls
#pragma unroll
    for (int kk = 0; kk < 2; ++kk) {
      const int kb = k0 + kk * 32;
      // A fragment: lane holds row `row`, K = kb+half*16..+15
      union { v16us v; v8us h[2]; } au;
      const unsigned short* ap = A + (size_t)row * K + kb + half * 16;
      au.h[0] = *(const v8us*)(ap);
      au.h[1] = *(const v8us*)(ap + 8);
      v16bf a = __builtin_bit_cast(v16bf, au.v);

      const int t = kb >> 5;
#pragma unroll
      for (int q = 0; q < Q; ++q) {
        v16us bu = *(const v16us*)(Bp + (((size_t)t * ntiles + nt0 + q) * 32 + l) * 16);
        v16bf b = __builtin_bit_cast(v16bf, bu);
        acc[q] = __builtin_amdgcn_wmma_f32_16x16x32_bf16(false, a, false, b,
                                                         (short)0, acc[q], false, false);
      }
    }
  }
#pragma unroll
  for (int q = 0; q < Q; ++q) {
    const int ncol = ((nt0 + q) << 4) + lane;
    const float badd = bias ? bias[ncol] : 0.0f;
#pragma unroll
    for (int i = 0; i < 8; ++i) {
      int r = m0 + i + half * 8;
      C[(size_t)r * N + ncol] = acc[q][i] + badd;
    }
  }
}

// ---------------- per-node attention logits: as/ad [NN,HEADS] ----------------
__global__ void attn_logits(const float* __restrict__ xp,
                            const float* __restrict__ a_src,
                            const float* __restrict__ a_dst,
                            float* __restrict__ as_, float* __restrict__ ad_) {
  int idx = blockIdx.x * blockDim.x + threadIdx.x;   // n*HEADS + h
  if (idx >= NN * HEADS) return;
  int h = idx & (HEADS - 1);
  const float* xr = xp + (size_t)(idx >> 2) * HC + h * CH;
  const float* wsp = a_src + h * CH;
  const float* wdp = a_dst + h * CH;
  float s = 0.f, d = 0.f;
#pragma unroll 4
  for (int c = 0; c < CH; ++c) { float v = xr[c]; s += v * wsp[c]; d += v * wdp[c]; }
  as_[idx] = s; ad_[idx] = d;
}

// ---------------- edge pass 1: segment max of leaky(e) over targets ----------------
__global__ void edge_max(const int* __restrict__ ei,
                         const float* __restrict__ as_, const float* __restrict__ ad_,
                         float* __restrict__ mx) {
  int e = blockIdx.x * blockDim.x + threadIdx.x;
  if (e >= ET) return;
  int src = (e < NE) ? ei[e]      : (e - NE);
  int tgt = (e < NE) ? ei[NE + e] : (e - NE);
#pragma unroll
  for (int h = 0; h < HEADS; ++h) {
    float v = as_[src * HEADS + h] + ad_[tgt * HEADS + h];
    v = (v > 0.f) ? v : v * NEG;
    atomMaxF(&mx[tgt * HEADS + h], v);
  }
}

// ---------------- edge pass 2: exp + segment sum ----------------
__global__ void edge_expsum(const int* __restrict__ ei,
                            const float* __restrict__ as_, const float* __restrict__ ad_,
                            const float* __restrict__ mx,
                            float* __restrict__ ex, float* __restrict__ den) {
  int e = blockIdx.x * blockDim.x + threadIdx.x;
  if (e >= ET) return;
  int src = (e < NE) ? ei[e]      : (e - NE);
  int tgt = (e < NE) ? ei[NE + e] : (e - NE);
#pragma unroll
  for (int h = 0; h < HEADS; ++h) {
    float v = as_[src * HEADS + h] + ad_[tgt * HEADS + h];
    v = (v > 0.f) ? v : v * NEG;
    float x = __expf(v - mx[tgt * HEADS + h]);
    ex[(size_t)e * HEADS + h] = x;
    atomicAdd(&den[tgt * HEADS + h], x);
  }
}

// ---------------- edge pass 3: weighted aggregation (+head mean 1/4) ----------------
// 256 threads = 4 edges x 64 channels; alphas staged once in LDS per block.
__global__ __launch_bounds__(256)
void edge_aggr(const int* __restrict__ ei,
               const float* __restrict__ xp,
               const float* __restrict__ ex, const float* __restrict__ den,
               float* __restrict__ out) {
  __shared__ float alpha[4][HEADS];
  __shared__ int   ssrc[4], stgt[4];
  const int  t  = threadIdx.x;
  const long e0 = (long)blockIdx.x * 4;
  if (t < 16) {
    long ee = e0 + (t >> 2);
    int  h  = t & 3;
    if (ee < ET) {
      int sr = (ee < NE) ? ei[ee]      : (int)(ee - NE);
      int tg = (ee < NE) ? ei[NE + ee] : (int)(ee - NE);
      alpha[t >> 2][h] = ex[(size_t)ee * HEADS + h] / (den[tg * HEADS + h] + 1e-16f);
      if (h == 0) { ssrc[t >> 2] = sr; stgt[t >> 2] = tg; }
    }
  }
  __syncthreads();
  const int  le = t >> 6;        // local edge 0..3
  const int  c  = t & 63;
  const long e  = e0 + le;
  if (e >= ET) return;
  const float* xr = xp + (size_t)ssrc[le] * HC + c;
  float acc = alpha[le][0] * xr[0]       + alpha[le][1] * xr[CH]
            + alpha[le][2] * xr[2 * CH]  + alpha[le][3] * xr[3 * CH];
  atomicAdd(&out[(size_t)stgt[le] * CH + c], 0.25f * acc);
}

// ---------------- BatchNorm stats: stats[0..C)=sum, stats[C..2C)=sumsq ----------------
__global__ void bn_stats(const float* __restrict__ x, int n, int rows_per,
                         float* __restrict__ stats) {
  int C = blockDim.x;
  int c = threadIdx.x;
  int r0 = blockIdx.x * rows_per;
  int r1 = min(r0 + rows_per, n);
  float s = 0.f, q = 0.f;
  for (int r = r0; r < r1; ++r) { float v = x[(size_t)r * C + c]; s += v; q += v * v; }
  atomicAdd(&stats[c], s);
  atomicAdd(&stats[C + c], q);
}

// ---------------- BatchNorm apply (+ReLU) ----------------
__global__ void bn_apply(const float* __restrict__ x, float* __restrict__ y,
                         const float* __restrict__ stats,
                         const float* __restrict__ g, const float* __restrict__ b,
                         long total, int C, float inv_n) {
  long i = (long)blockIdx.x * blockDim.x + threadIdx.x;
  if (i >= total) return;
  int c = (int)(i % C);
  float mu  = stats[c] * inv_n;
  float var = stats[C + c] * inv_n - mu * mu;
  float v = (x[i] - mu) * rsqrtf(var + EPSBN) * g[c] + b[c];
  y[i] = v > 0.f ? v : 0.f;
}

// ---------------- pooling score: s[n] = sum_j tanh(T[n,j]+b1[j])*W2[j] + b2 ----------------
__global__ __launch_bounds__(256)
void pool_score(const float* __restrict__ T, const float* __restrict__ b1,
                const float* __restrict__ W2, const float* __restrict__ b2,
                float* __restrict__ s) {
  __shared__ float red[256];
  int n = blockIdx.x, j = threadIdx.x;
  red[j] = tanhf(T[(size_t)n * MLPH + j] + b1[j]) * W2[j];
  __syncthreads();
  for (int off = 128; off > 0; off >>= 1) {
    if (j < off) red[j] += red[j + off];
    __syncthreads();
  }
  if (j == 0) s[n] = red[0] + b2[0];
}

// ---------------- per-graph softmax pooling ----------------
__global__ void batch_max(const float* __restrict__ s, const int* __restrict__ bidx,
                          float* __restrict__ mb) {
  int n = blockIdx.x * blockDim.x + threadIdx.x;
  if (n >= NN) return;
  atomMaxF(&mb[bidx[n]], s[n]);
}
__global__ void batch_expsum(const float* __restrict__ s, const int* __restrict__ bidx,
                             const float* __restrict__ mb,
                             float* __restrict__ wv, float* __restrict__ db) {
  int n = blockIdx.x * blockDim.x + threadIdx.x;
  if (n >= NN) return;
  int b = bidx[n];
  float e = __expf(s[n] - mb[b]);
  wv[n] = e;
  atomicAdd(&db[b], e);
}
__global__ void pool_aggr(const float* __restrict__ h, const int* __restrict__ bidx,
                          const float* __restrict__ wv, const float* __restrict__ db,
                          float* __restrict__ G) {
  long i = (long)blockIdx.x * blockDim.x + threadIdx.x;
  if (i >= (long)NN * CH) return;
  int n = (int)(i >> 6), c = (int)(i & 63);
  int b = bidx[n];
  float w = wv[n] / (db[b] + 1e-16f);
  atomicAdd(&G[b * CH + c], h[(size_t)n * CH + c] * w);
}

// ================= host-side helpers =================
static inline void fill_async(float* p, long n, float v, hipStream_t s) {
  fill_kernel<<<(unsigned)((n + 255) / 256), 256, 0, s>>>(p, n, v);
}
static inline void run_gemm(const float* A, const float* B, const float* bias, float* C,
                            int M, int K, int N,
                            unsigned short* ABF, unsigned short* BPK, hipStream_t s) {
  long na = (long)M * K, nb = (long)K * N;
  cvt_bf16_kernel<<<(unsigned)((na + 255) / 256), 256, 0, s>>>(A, ABF, na);
  pack_B_kernel<<<(unsigned)((nb + 255) / 256), 256, 0, s>>>(B, BPK, K, N);
  const int ntiles = N >> 4;
  const int full = ntiles >> 2, rem = ntiles & 3;
  if (full)
    gemm_bf16_wmma<4><<<dim3(M / 16, full), 32, 0, s>>>(ABF, BPK, bias, C, M, K, N, 0);
  if (rem == 1)
    gemm_bf16_wmma<1><<<dim3(M / 16, 1), 32, 0, s>>>(ABF, BPK, bias, C, M, K, N, full * 4);
  else if (rem == 2)
    gemm_bf16_wmma<2><<<dim3(M / 16, 1), 32, 0, s>>>(ABF, BPK, bias, C, M, K, N, full * 4);
  else if (rem == 3)
    gemm_bf16_wmma<3><<<dim3(M / 16, 1), 32, 0, s>>>(ABF, BPK, bias, C, M, K, N, full * 4);
}

extern "C" void kernel_launch(void* const* d_in, const int* in_sizes, int n_in,
                              void* d_out, int out_size, void* d_ws, size_t ws_size,
                              hipStream_t stream) {
  // ---- inputs in setup_inputs() dict insertion order ----
  const float* x    = (const float*)d_in[0];
  const int*   ei   = (const int*)d_in[1];
  const int*   bidx = (const int*)d_in[2];
  const float* gW[3]  = {(const float*)d_in[3],  (const float*)d_in[7],  (const float*)d_in[11]};
  const float* gAS[3] = {(const float*)d_in[4],  (const float*)d_in[8],  (const float*)d_in[12]};
  const float* gAD[3] = {(const float*)d_in[5],  (const float*)d_in[9],  (const float*)d_in[13]};
  // gat biases d_in[6,10,14] cancel inside BatchNorm -> unused
  const float* bnG[3] = {(const float*)d_in[15], (const float*)d_in[17], (const float*)d_in[19]};
  const float* bnB[3] = {(const float*)d_in[16], (const float*)d_in[18], (const float*)d_in[20]};
  const float* a1W = (const float*)d_in[21];
  const float* a1b = (const float*)d_in[22];
  const float* a2W = (const float*)d_in[23];
  const float* a2b = (const float*)d_in[24];
  const float* p1W = (const float*)d_in[25];   // p1.b (26) cancels in BN
  const float* pbn1g = (const float*)d_in[27];
  const float* pbn1b = (const float*)d_in[28];
  const float* p2W = (const float*)d_in[29];   // p2.b (30) cancels in BN
  const float* pbn2g = (const float*)d_in[31];
  const float* pbn2b = (const float*)d_in[32];
  const float* p3W = (const float*)d_in[33];
  const float* p3b = (const float*)d_in[34];
  float* out = (float*)d_out;

  // ---- workspace layout (float offsets; bf16 buffers at the end, 32B aligned) ----
  float* ws = (float*)d_ws;
  float* XP   = ws;                                  // [NN,256]  (reused as pool-MLP tmp)
  float* H    = XP  + (size_t)NN * HC;               // [NN,64]
  float* OUT  = H   + (size_t)NN * CH;               // [NN,64]
  float* AS   = OUT + (size_t)NN * CH;               // [NN,4]
  float* AD   = AS  + (size_t)NN * HEADS;            // [NN,4]
  float* MX   = AD  + (size_t)NN * HEADS;            // [NN,4]
  float* DEN  = MX  + (size_t)NN * HEADS;            // [NN,4]
  float* EX   = DEN + (size_t)NN * HEADS;            // [ET,4]
  float* SC   = EX  + (size_t)ET * HEADS;            // [NN]
  float* WV   = SC  + NN;                            // [NN]
  float* ST   = WV  + NN;                            // stats [2*256]
  float* MB   = ST  + 2 * MLPH;                      // [64]
  float* DB   = MB  + BG;                            // [64]
  float* G    = DB  + BG;                            // [64,64]
  float* Z1   = G   + BG * CH;                       // [64,256]
  float* Z2   = Z1  + BG * MLPH;                     // [64,128]
  float* FEND = Z2  + BG * MLPH2;
  unsigned short* ABF = (unsigned short*)(FEND + 64);           // [NN*128] bf16 (max A)
  unsigned short* BPK = ABF + (size_t)NN * FIN + 64;            // [128*10000] bf16 (max B)

  const int eb = 256;
  const unsigned egrid = (ET + eb - 1) / eb;

  // ================= GAT layers =================
  const float* hin = x;
  int kdim = FIN;
  for (int l = 0; l < 3; ++l) {
    // xp = h @ W  -> [NN, 256]   (WMMA bf16, pre-converted operands)
    run_gemm(hin, gW[l], nullptr, XP, NN, kdim, HC, ABF, BPK, stream);
    // per-node attention logits
    attn_logits<<<(NN * HEADS + 255) / 256, 256, 0, stream>>>(XP, gAS[l], gAD[l], AS, AD);
    // init segment buffers
    fill_async(MX,  (long)NN * HEADS, -3.4e38f, stream);
    fill_async(DEN, (long)NN * HEADS, 0.f, stream);
    fill_async(OUT, (long)NN * CH,    0.f, stream);
    // segment softmax over in-edges + aggregation
    edge_max   <<<egrid, eb, 0, stream>>>(ei, AS, AD, MX);
    edge_expsum<<<egrid, eb, 0, stream>>>(ei, AS, AD, MX, EX, DEN);
    edge_aggr  <<<(unsigned)((ET + 3) / 4), 256, 0, stream>>>(ei, XP, EX, DEN, OUT);
    // BatchNorm + ReLU  (gat bias cancels in BN)
    fill_async(ST, 2 * CH, 0.f, stream);
    bn_stats<<<(NN + 127) / 128, CH, 0, stream>>>(OUT, NN, 128, ST);
    {
      long tot = (long)NN * CH;
      bn_apply<<<(unsigned)((tot + 255) / 256), 256, 0, stream>>>(OUT, H, ST, bnG[l], bnB[l],
                                                                  tot, CH, 1.0f / NN);
    }
    hin = H;
    kdim = CH;
  }

  // ================= attention pooling =================
  // T = h @ attn1.W  -> [NN,256] (reuse XP)
  run_gemm(H, a1W, nullptr, XP, NN, CH, MLPH, ABF, BPK, stream);
  pool_score<<<NN, MLPH, 0, stream>>>(XP, a1b, a2W, a2b, SC);
  fill_async(MB, BG, -3.4e38f, stream);
  fill_async(DB, BG, 0.f, stream);
  fill_async(G,  BG * CH, 0.f, stream);
  batch_max   <<<(NN + 255) / 256, 256, 0, stream>>>(SC, bidx, MB);
  batch_expsum<<<(NN + 255) / 256, 256, 0, stream>>>(SC, bidx, MB, WV, DB);
  pool_aggr<<<(unsigned)(((long)NN * CH + 255) / 256), 256, 0, stream>>>(H, bidx, WV, DB, G);

  // ================= MLP head =================
  // z1 = relu(bn(G @ p1.W))   (p1.b cancels in BN)
  run_gemm(G, p1W, nullptr, Z1, BG, CH, MLPH, ABF, BPK, stream);
  fill_async(ST, 2 * MLPH, 0.f, stream);
  bn_stats<<<1, MLPH, 0, stream>>>(Z1, BG, BG, ST);
  bn_apply<<<(BG * MLPH + 255) / 256, 256, 0, stream>>>(Z1, Z1, ST, pbn1g, pbn1b,
                                                        (long)BG * MLPH, MLPH, 1.0f / BG);
  // z2 = relu(bn(z1 @ p2.W))  (p2.b cancels in BN)
  run_gemm(Z1, p2W, nullptr, Z2, BG, MLPH, MLPH2, ABF, BPK, stream);
  fill_async(ST, 2 * MLPH2, 0.f, stream);
  bn_stats<<<1, MLPH2, 0, stream>>>(Z2, BG, BG, ST);
  bn_apply<<<(BG * MLPH2 + 255) / 256, 256, 0, stream>>>(Z2, Z2, ST, pbn2g, pbn2b,
                                                         (long)BG * MLPH2, MLPH2, 1.0f / BG);
  // out = z2 @ p3.W + p3.b    -> [64, 10000]
  run_gemm(Z2, p3W, p3b, out, BG, MLPH2, NGLOB, ABF, BPK, stream);
}
